// FourierAttention_27754078666803
// MI455X (gfx1250) — compile-verified, matrix operand order checked by reading, not run on hardware
//
#include <hip/hip_runtime.h>
#include <cstdint>
#include <cstddef>

// ---------------------------------------------------------------------------
// Types for CDNA5 WMMA (wave32): A/B = 16 x bf16 per lane, C/D = 8 x f32.
// ---------------------------------------------------------------------------
typedef __attribute__((ext_vector_type(16))) __bf16 v16bf;
typedef __attribute__((ext_vector_type(8)))  __bf16 v8bf;
typedef __attribute__((ext_vector_type(8)))  float  v8f;

// fp32 -> bf16 with round-to-nearest-even
__device__ __forceinline__ unsigned short f2bf(float f) {
  unsigned int u = __float_as_uint(f);
  unsigned int r = u + 0x7fffu + ((u >> 16) & 1u);
  return (unsigned short)(r >> 16);
}

// Load one lane's 16 bf16 fragment elements per ISA 7.12.2 16-bit layout:
// two contiguous 8-element (16-byte) K chunks at p and p+16 elements.
__device__ __forceinline__ v16bf load2(const unsigned short* __restrict__ p) {
  v8bf c0 = *reinterpret_cast<const v8bf*>(p);
  v8bf c1 = *reinterpret_cast<const v8bf*>(p + 16);
  v16bf f;
#pragma unroll
  for (int i = 0; i < 8; ++i) { f[i] = c0[i]; f[i + 8] = c1[i]; }
  return f;
}

#define WMMA_BF16(Af, Bf, Cf) \
  __builtin_amdgcn_wmma_f32_16x16x32_bf16(false, (Af), false, (Bf), (short)0, (Cf), false, false)

// ---------------------------------------------------------------------------
// Generic GEMM:  C[M,N] = alpha * (A[M,K] . B[N,K]^T) + bias[N]
// A, B bf16 row-major; C fp32 or bf16.
// Each wave computes a 64x32 tile (4x2 of 16x16 WMMA tiles): per K=32 step,
// 6 fragment loads (12 x b128) feed 8 v_wmma. Fragments are software-
// pipelined one K-step ahead so WMMAs never wait on just-issued loads.
// Grid must supply exactly (M/64)*(N/32) waves.  K must be a multiple of 32
// and >= 64.
// ---------------------------------------------------------------------------
template <bool OUT_BF16>
__global__ void gemm_wmma_bf16(const unsigned short* __restrict__ A, int lda,
                               const unsigned short* __restrict__ Bm, int ldb,
                               void* __restrict__ Cv, int ldc,
                               const float* __restrict__ bias,
                               float alpha, int N, int K) {
  const int wave    = (int)((blockIdx.x * blockDim.x + threadIdx.x) >> 5);
  const int tiles_n = N >> 5;           // 32-wide N tiles
  const int tm = (wave / tiles_n) << 6; // 64-tall M tile
  const int tn = (wave % tiles_n) << 5;

  const int lane = threadIdx.x & 31;
  const int l15  = lane & 15;
  const int hi   = lane >> 4;

  // Running element offsets off the global-argument bases (keeps global_load
  // addressing; subtile rows reached via constant 16*ld strides).
  size_t offA = (size_t)(tm + l15) * (size_t)lda + (size_t)(hi * 8);
  size_t offB = (size_t)(tn + l15) * (size_t)ldb + (size_t)(hi * 8);
  const size_t sA = (size_t)16 * (size_t)lda;
  const size_t sB = (size_t)16 * (size_t)ldb;

  v8f c00 = {}, c01 = {};
  v8f c10 = {}, c11 = {};
  v8f c20 = {}, c21 = {};
  v8f c30 = {}, c31 = {};

  // ---- prologue: load K-step 0 fragments ----
  v16bf b0 = load2(Bm + offB);
  v16bf b1 = load2(Bm + offB + sB);
  v16bf a0 = load2(A + offA);
  v16bf a1 = load2(A + offA + sA);
  v16bf a2 = load2(A + offA + 2 * sA);
  v16bf a3 = load2(A + offA + 3 * sA);

  // ---- steady state: issue loads for step k, compute step k-1 ----
  for (int k0 = 32; k0 < K; k0 += 32) {
    offA += 32;
    offB += 32;

    // prefetch A rows one further K-step ahead
    __builtin_prefetch(A + offA + 32, 0, 3);
    __builtin_prefetch(A + offA + sA + 32, 0, 3);
    __builtin_prefetch(A + offA + 2 * sA + 32, 0, 3);
    __builtin_prefetch(A + offA + 3 * sA + 32, 0, 3);

    v16bf nb0 = load2(Bm + offB);
    v16bf nb1 = load2(Bm + offB + sB);
    v16bf na0 = load2(A + offA);
    v16bf na1 = load2(A + offA + sA);
    v16bf na2 = load2(A + offA + 2 * sA);
    v16bf na3 = load2(A + offA + 3 * sA);

    c00 = WMMA_BF16(a0, b0, c00);
    c01 = WMMA_BF16(a0, b1, c01);
    c10 = WMMA_BF16(a1, b0, c10);
    c11 = WMMA_BF16(a1, b1, c11);
    c20 = WMMA_BF16(a2, b0, c20);
    c21 = WMMA_BF16(a2, b1, c21);
    c30 = WMMA_BF16(a3, b0, c30);
    c31 = WMMA_BF16(a3, b1, c31);

    a0 = na0; a1 = na1; a2 = na2; a3 = na3;
    b0 = nb0; b1 = nb1;
  }

  // ---- epilogue: last K-step ----
  c00 = WMMA_BF16(a0, b0, c00);
  c01 = WMMA_BF16(a0, b1, c01);
  c10 = WMMA_BF16(a1, b0, c10);
  c11 = WMMA_BF16(a1, b1, c11);
  c20 = WMMA_BF16(a2, b0, c20);
  c21 = WMMA_BF16(a2, b1, c21);
  c30 = WMMA_BF16(a3, b0, c30);
  c31 = WMMA_BF16(a3, b1, c31);

  // C/D layout: VGPR r holds row (r + hi*8) of the 16x16 tile, lane l15 = col.
  v8f acc[4][2] = {{c00, c01}, {c10, c11}, {c20, c21}, {c30, c31}};
#pragma unroll
  for (int mi = 0; mi < 4; ++mi) {
#pragma unroll
    for (int ni = 0; ni < 2; ++ni) {
      const int col = tn + ni * 16 + l15;
      const float bv = bias ? bias[col] : 0.0f;
      v8f a = acc[mi][ni];
#pragma unroll
      for (int r = 0; r < 8; ++r) {
        const int row = tm + mi * 16 + hi * 8 + r;
        float v = a[r] * alpha + bv;
        if constexpr (OUT_BF16)
          reinterpret_cast<unsigned short*>(Cv)[(size_t)row * (size_t)ldc + col] = f2bf(v);
        else
          reinterpret_cast<float*>(Cv)[(size_t)row * (size_t)ldc + col] = v;
      }
    }
  }
}

// ---------------------------------------------------------------------------
// fp32 -> bf16 conversion, 4 elements / thread (b128 in, b64 out)
// ---------------------------------------------------------------------------
__global__ void cvt_f32_bf16(const float* __restrict__ in,
                             unsigned short* __restrict__ out, int n) {
  int i = (int)(blockIdx.x * blockDim.x + threadIdx.x) * 4;
  if (i + 3 < n) {
    float4 f = *reinterpret_cast<const float4*>(in + i);
    unsigned long long pack =
        (unsigned long long)f2bf(f.x) |
        ((unsigned long long)f2bf(f.y) << 16) |
        ((unsigned long long)f2bf(f.z) << 32) |
        ((unsigned long long)f2bf(f.w) << 48);
    *reinterpret_cast<unsigned long long*>(out + i) = pack;
  }
}

// ---------------------------------------------------------------------------
// bf16 transpose [S,D] -> [D,S] per batch (blockIdx.z), LDS 32x33 tile
// ---------------------------------------------------------------------------
__global__ void transpose_bf16(const unsigned short* __restrict__ V,
                               unsigned short* __restrict__ Vt, int S, int D) {
  __shared__ unsigned short tile[32][33];
  const int b  = blockIdx.z;
  const int d0 = blockIdx.x * 32;
  const int s0 = blockIdx.y * 32;
  const int tx = threadIdx.x;   // 0..31
  const int ty = threadIdx.y;   // 0..7
  const unsigned short* Vb  = V  + (size_t)b * S * D;
  unsigned short*       Vtb = Vt + (size_t)b * S * D;
#pragma unroll
  for (int i = 0; i < 32; i += 8)
    tile[ty + i][tx] = Vb[(size_t)(s0 + ty + i) * D + d0 + tx];
  __syncthreads();
#pragma unroll
  for (int i = 0; i < 32; i += 8)
    Vtb[(size_t)(d0 + ty + i) * S + s0 + tx] = tile[tx][ty + i];
}

// ---------------------------------------------------------------------------
// Row softmax over ncols fp32 (ncols == 2048), bf16 probabilities out.
// One 256-thread block per row, 8 elements per thread.
// ---------------------------------------------------------------------------
__global__ void softmax_rows(const float* __restrict__ S,
                             unsigned short* __restrict__ P, int ncols) {
  __shared__ float red[256];
  const int row = blockIdx.x;
  const int t   = threadIdx.x;
  const float* r = S + (size_t)row * ncols;

  float local[8];
  float mx = -3.402823466e+38f;
#pragma unroll
  for (int i = 0; i < 8; ++i) {
    local[i] = r[t + i * 256];
    mx = fmaxf(mx, local[i]);
  }
  red[t] = mx; __syncthreads();
  for (int s = 128; s > 0; s >>= 1) {
    if (t < s) red[t] = fmaxf(red[t], red[t + s]);
    __syncthreads();
  }
  mx = red[0]; __syncthreads();

  float sum = 0.0f;
#pragma unroll
  for (int i = 0; i < 8; ++i) {
    local[i] = __expf(local[i] - mx);
    sum += local[i];
  }
  red[t] = sum; __syncthreads();
  for (int s = 128; s > 0; s >>= 1) {
    if (t < s) red[t] += red[t + s];
    __syncthreads();
  }
  const float inv = 1.0f / red[0];

  unsigned short* pr = P + (size_t)row * ncols;
#pragma unroll
  for (int i = 0; i < 8; ++i)
    pr[t + i * 256] = f2bf(local[i] * inv);
}

// ---------------------------------------------------------------------------
// Host orchestration
// ---------------------------------------------------------------------------
extern "C" void kernel_launch(void* const* d_in, const int* in_sizes, int n_in,
                              void* d_out, int out_size, void* d_ws, size_t ws_size,
                              hipStream_t stream) {
  (void)in_sizes; (void)n_in; (void)out_size; (void)ws_size;

  constexpr int B = 4, S = 2048, D = 1024;
  constexpr int M = B * S;                // 8192 total rows
  constexpr float SCALE = 0.125f;         // (1024/16)^-0.5

  const float* x  = (const float*)d_in[0];
  const float* Wq = (const float*)d_in[1];
  const float* bq = (const float*)d_in[2];
  const float* Wk = (const float*)d_in[3];
  const float* bk = (const float*)d_in[4];
  const float* Wv = (const float*)d_in[5];
  const float* bv = (const float*)d_in[6];
  const float* Wo = (const float*)d_in[7];
  const float* bo = (const float*)d_in[8];

  char* base = (char*)d_ws;
  size_t off = 0;
  auto alloc = [&](size_t bytes) -> void* {
    void* p = base + off;
    off = (off + bytes + 255) & ~(size_t)255;
    return p;
  };
  unsigned short* xb   = (unsigned short*)alloc((size_t)M * D * 2);   // 16 MB
  unsigned short* Wqb  = (unsigned short*)alloc((size_t)D * D * 2);   //  2 MB
  unsigned short* Wkb  = (unsigned short*)alloc((size_t)D * D * 2);
  unsigned short* Wvb  = (unsigned short*)alloc((size_t)D * D * 2);
  unsigned short* Wob  = (unsigned short*)alloc((size_t)D * D * 2);
  unsigned short* Qb   = (unsigned short*)alloc((size_t)M * D * 2);   // 16 MB
  unsigned short* Kb   = (unsigned short*)alloc((size_t)M * D * 2);
  unsigned short* Vb   = (unsigned short*)alloc((size_t)M * D * 2);
  unsigned short* Vt   = (unsigned short*)alloc((size_t)M * D * 2);   // [B,D,S]
  float*          Sbuf = (float*)alloc((size_t)S * S * 4);            // 16 MB, 1 batch
  unsigned short* Pbuf = (unsigned short*)alloc((size_t)S * S * 2);   //  8 MB, 1 batch
  unsigned short* Ob   = (unsigned short*)alloc((size_t)M * D * 2);   // 16 MB

  // --- fp32 -> bf16 conversions ---
  cvt_f32_bf16<<<(M * D / 4) / 256, 256, 0, stream>>>(x,  xb,  M * D);
  cvt_f32_bf16<<<(D * D / 4) / 256, 256, 0, stream>>>(Wq, Wqb, D * D);
  cvt_f32_bf16<<<(D * D / 4) / 256, 256, 0, stream>>>(Wk, Wkb, D * D);
  cvt_f32_bf16<<<(D * D / 4) / 256, 256, 0, stream>>>(Wv, Wvb, D * D);
  cvt_f32_bf16<<<(D * D / 4) / 256, 256, 0, stream>>>(Wo, Wob, D * D);

  // (M/64)*(N/32) waves, 4 waves (128 threads) per block
  auto gemm_blocks = [](int m, int n) { return ((m / 64) * (n / 32)) / 4; };

  // --- Q/K/V projections: [M,D] = xb . W^T + b ---
  gemm_wmma_bf16<true><<<gemm_blocks(M, D), 128, 0, stream>>>(
      xb, D, Wqb, D, Qb, D, bq, 1.0f, D, D);
  gemm_wmma_bf16<true><<<gemm_blocks(M, D), 128, 0, stream>>>(
      xb, D, Wkb, D, Kb, D, bk, 1.0f, D, D);
  gemm_wmma_bf16<true><<<gemm_blocks(M, D), 128, 0, stream>>>(
      xb, D, Wvb, D, Vb, D, bv, 1.0f, D, D);

  // --- V^T so that P.V B-operand loads are K-contiguous ---
  {
    dim3 g(D / 32, S / 32, B), blk(32, 8, 1);
    transpose_bf16<<<g, blk, 0, stream>>>(Vb, Vt, S, D);
  }

  // --- attention per batch (score buffer reused; serial on stream) ---
  for (int b = 0; b < B; ++b) {
    const unsigned short* Qbb = Qb + (size_t)b * S * D;
    const unsigned short* Kbb = Kb + (size_t)b * S * D;
    const unsigned short* Vtb = Vt + (size_t)b * S * D;   // [D,S]
    unsigned short*       Obb = Ob + (size_t)b * S * D;

    // scores[S,S] = SCALE * Q . K^T   (fp32 out)
    gemm_wmma_bf16<false><<<gemm_blocks(S, S), 128, 0, stream>>>(
        Qbb, D, Kbb, D, Sbuf, S, nullptr, SCALE, S, D);

    // softmax rows -> bf16 probabilities
    softmax_rows<<<S, 256, 0, stream>>>(Sbuf, Pbuf, S);

    // O[S,D] = P . V   (B-operand = Vt rows, ldb = S)
    gemm_wmma_bf16<true><<<gemm_blocks(S, D), 128, 0, stream>>>(
        Pbuf, S, Vtb, S, Obb, D, nullptr, 1.0f, D, S);
  }

  // --- final projection: out[M,D] = O . Wo^T + bo  (fp32 out) ---
  gemm_wmma_bf16<false><<<gemm_blocks(M, D), 128, 0, stream>>>(
      Ob, D, Wob, D, d_out, D, bo, 1.0f, D, D);
}
